// RNN_90829968376262
// MI455X (gfx1250) — matrix operand order repaired and checked by reading, not compiled, not run on hardware
//
#include <hip/hip_runtime.h>
#include <cmath>

// ---------- vector types ----------
typedef __attribute__((ext_vector_type(16))) _Float16 v16h;
typedef __attribute__((ext_vector_type(8)))  _Float16 h8;
typedef __attribute__((ext_vector_type(8)))  float    v8f;

#define BM 128
#define BN 64
#define BK 32
#define PADK 40   // halfs per LDS row (80B): 16 consecutive rows hit distinct banks
#define NSTAGE 3  // async pipeline depth (Q >= issue_distance + 2 for WAR safety)

// ---- CDNA5 async global->LDS copy (ASYNCcnt-tracked), GV addressing ----
__device__ __forceinline__ unsigned lds_off(const void* p) {
  // generic pointers to LDS carry the LDS byte offset in the low 32 bits
  return (unsigned)(uintptr_t)p;
}
__device__ __forceinline__ void async_copy_b128(unsigned lds, const void* g) {
  asm volatile("global_load_async_to_lds_b128 %0, %1, off"
               :: "v"(lds), "v"((unsigned long long)(uintptr_t)g)
               : "memory");
}
__device__ __forceinline__ void wait_async_le(int n) {
  if (n == 0) asm volatile("s_wait_asynccnt 0x0" ::: "memory");
  else        asm volatile("s_wait_asynccnt 0x3" ::: "memory");
}

// C[M,N] = op(A[M,K] * BT[N,K]^T), A/BT f16, f32 WMMA accumulate.
// Epilogue: optional +addsrc(f16), +bias(f32 per column), tanh; C f16 or f32.
template<bool TANH, bool BIAS, bool ADD, bool OUTH>
__global__ __launch_bounds__(256) void gemm_f16_async(
    const _Float16* __restrict__ A,    // [M][K] row-major f16
    const _Float16* __restrict__ BT,   // [N][K] (B transposed) f16
    const float* __restrict__ bias,    // [N]
    const _Float16* addsrc,            // [M][N] f16, used iff ADD
    void* Cout, int M, int N, int K)
{
  __shared__ _Float16 As[NSTAGE][BM * PADK];   // [m][k] ring
  __shared__ _Float16 Bs[NSTAGE][BN * PADK];   // [n][k] ring

  const int tid  = threadIdx.x;
  const int lane = tid & 31;
  const int wv   = tid >> 5;
  const int hl   = lane >> 4;     // lane half
  const int l15  = lane & 15;

  const int m0 = blockIdx.y * BM;
  const int n0 = blockIdx.x * BN;

  const int mt = (wv >> 1) * 32;  // wave M offset: 0,32,64,96
  const int nt = (wv & 1) * 32;   // wave N offset: 0,32

  // Each 16-byte chunk = 8 halfs. A tile: 128 rows x 4 chunks (2/thread).
  // B tile: 64 rows x 4 chunks (1/thread). => 3 async ops / thread / tile.
  auto issue_tile = [&](int kk, int s) {
#pragma unroll
    for (int i = 0; i < 2; ++i) {
      const int ca = tid + i * 256;
      const int r  = ca >> 2;
      const int c  = (ca & 3) * 8;
      async_copy_b128(lds_off(&As[s][r * PADK + c]),
                      A + (long)(m0 + r) * K + kk + c);
    }
    const int rb = tid >> 2;
    const int cb = (tid & 3) * 8;
    async_copy_b128(lds_off(&Bs[s][rb * PADK + cb]),
                    BT + (long)(n0 + rb) * K + kk + cb);
  };

  v8f acc[2][2] = {};

  issue_tile(0, 0);               // tile 0 -> slot 0

  int p = 0;
  for (int k0 = 0; k0 < K; k0 += BK) {
    const int pn = (p + 1 == NSTAGE) ? 0 : p + 1;
    if (k0 + BK < K) {
      issue_tile(k0 + BK, pn);    // tile i+1 in flight during tile i compute
      wait_async_le(3);           // oldest 3 (tile i) landed; newest 3 may fly
    } else {
      wait_async_le(0);           // last tile: drain everything
    }
    __syncthreads();              // tile i visible to all waves

    // ---- A fragments (16x32 f16; elems 0..7 <- K=8*hl+i, 8..15 <- K=16+8*hl+i)
    v16h afr[2];
#pragma unroll
    for (int i = 0; i < 2; ++i) {
      const _Float16* ap = &As[p][(mt + i * 16 + l15) * PADK + 8 * hl];
      h8 lo = *(const h8*)(ap);
      h8 hi = *(const h8*)(ap + 16);
#pragma unroll
      for (int j = 0; j < 8; ++j) { afr[i][j] = lo[j]; afr[i][j + 8] = hi[j]; }
    }
    // ---- B fragments (32x16 f16; elems 0..7 <- K=8*hl+i, 8..15 <- K=16+8*hl+i at column n)
    v16h bfr[2];
#pragma unroll
    for (int i = 0; i < 2; ++i) {
      const _Float16* bp = &Bs[p][(nt + i * 16 + l15) * PADK + 8 * hl];
      h8 lo = *(const h8*)(bp);
      h8 hi = *(const h8*)(bp + 16);
#pragma unroll
      for (int j = 0; j < 8; ++j) { bfr[i][j] = lo[j]; bfr[i][j + 8] = hi[j]; }
    }

#pragma unroll
    for (int i = 0; i < 2; ++i)
#pragma unroll
      for (int j = 0; j < 2; ++j)
        acc[i][j] = __builtin_amdgcn_wmma_f32_16x16x32_f16(
            false, afr[i], false, bfr[j], (short)0, acc[i][j], false, false);

    p = pn;
  }

  // ---- epilogue: C/D layout VGPR r -> M = r + 8*hl, N = l15 ----
#pragma unroll
  for (int i = 0; i < 2; ++i)
#pragma unroll
    for (int j = 0; j < 2; ++j) {
      const int ng = n0 + nt + j * 16 + l15;
#pragma unroll
      for (int r = 0; r < 8; ++r) {
        const int mg = m0 + mt + i * 16 + r + 8 * hl;
        float v = acc[i][j][r];
        if constexpr (ADD)  v += (float)addsrc[(long)mg * N + ng];
        if constexpr (BIAS) v += bias[ng];
        if constexpr (TANH) v = tanhf(v);
        if constexpr (OUTH) ((_Float16*)Cout)[(long)mg * N + ng] = (_Float16)v;
        else                ((float*)Cout)[(long)mg * N + ng]    = v;
      }
    }
}

// ---------- small prep / epilogue kernels ----------
__global__ void cvt_f32_to_f16_kernel(const float* s, _Float16* d, int n) {
  int i = blockIdx.x * 256 + threadIdx.x;
  if (i < n) d[i] = (_Float16)s[i];
}
// W[K][N] f32 -> WT[N][K] f16
__global__ void transpose_cvt_kernel(const float* W, _Float16* WT, int K, int N) {
  int i = blockIdx.x * 256 + threadIdx.x;
  if (i < K * N) {
    int k = i / N, n = i - k * N;
    WT[(long)n * K + k] = (_Float16)W[i];
  }
}
__global__ void tanh_f16_kernel(_Float16* p, int n) {
  int i = blockIdx.x * 256 + threadIdx.x;
  if (i < n) p[i] = (_Float16)tanhf((float)p[i]);
}
__global__ void cvt_f16_to_f32_kernel(const _Float16* s, float* d, int n) {
  int i = blockIdx.x * 256 + threadIdx.x;
  if (i < n) d[i] = (float)s[i];
}

extern "C" void kernel_launch(void* const* d_in, const int* in_sizes, int n_in,
                              void* d_out, int out_size, void* d_ws, size_t ws_size,
                              hipStream_t stream) {
  constexpr int T = 256, Bb = 256, V = 512, H = 1024;

  const float* inputs  = (const float*)d_in[0];
  const float* W_xh    = (const float*)d_in[1];
  const float* W_hh    = (const float*)d_in[2];
  const float* b_h     = (const float*)d_in[3];
  const float* W_dense = (const float*)d_in[4];
  const float* b_dense = (const float*)d_in[5];
  float* out = (float*)d_out;

  // workspace layout (all f16): inputs | xh/hs (in place) | WxhT | WhhT | WdT
  _Float16* x16  = (_Float16*)d_ws;                 // [T*B*V]
  _Float16* hx   = x16  + (long)T * Bb * V;         // [T*B*H]  xh -> hs in place
  _Float16* WxhT = hx   + (long)T * Bb * H;         // [H][V]
  _Float16* WhhT = WxhT + (long)H * V;              // [H][H]
  _Float16* WdT  = WhhT + (long)H * H;              // [V][H]

  dim3 blk(256);

  // prep: convert activations, convert+transpose weights to f16
  cvt_f32_to_f16_kernel<<<(T * Bb * V) / 256, blk, 0, stream>>>(inputs, x16, T * Bb * V);
  transpose_cvt_kernel<<<(V * H + 255) / 256, blk, 0, stream>>>(W_xh,    WxhT, V, H);
  transpose_cvt_kernel<<<(H * H + 255) / 256, blk, 0, stream>>>(W_hh,    WhhT, H, H);
  transpose_cvt_kernel<<<(H * V + 255) / 256, blk, 0, stream>>>(W_dense, WdT,  H, V);

  // Phase 1: xh = inputs @ W_xh + b_h    (M=T*B, N=H, K=V) -> f16
  gemm_f16_async<false, true, false, true>
      <<<dim3(H / BN, (T * Bb) / BM), blk, 0, stream>>>(
          x16, WxhT, b_h, nullptr, hx, T * Bb, H, V);

  // t = 0: hs[0] = tanh(xh[0])
  tanh_f16_kernel<<<(Bb * H + 255) / 256, blk, 0, stream>>>(hx, Bb * H);

  // t = 1..T-1: hs[t] = tanh(xh[t] + hs[t-1] @ W_hh)   (M=B, N=H, K=H)
  for (int t = 1; t < T; ++t) {
    const _Float16* hprev = hx + (long)(t - 1) * Bb * H;
    _Float16*       cur   = hx + (long)t * Bb * H;
    gemm_f16_async<true, false, true, true>
        <<<dim3(H / BN, Bb / BM), blk, 0, stream>>>(
            hprev, WhhT, nullptr, cur, cur, Bb, H, H);
  }

  // Phase 3: out = hs @ W_dense + b_dense  (M=T*B, N=V, K=H) -> f32
  gemm_f16_async<false, true, false, false>
      <<<dim3(V / BN, (T * Bb) / BM), blk, 0, stream>>>(
          hx, WdT, b_dense, nullptr, out, T * Bb, V, H);

  // Final state = (f32) hs[T-1], appended after outputs
  cvt_f16_to_f32_kernel<<<(Bb * H + 255) / 256, blk, 0, stream>>>(
      hx + (long)(T - 1) * Bb * H, out + (long)T * Bb * V, Bb * H);
}
